// GRU_72129680769130
// MI455X (gfx1250) — matrix operand order repaired and checked by reading, not compiled, not run on hardware
//
#include <hip/hip_runtime.h>

// ---------------------------------------------------------------------------
// Fused 2-layer GRU + MLP head for MI455X (gfx1250, wave32, WMMA bf16)
// B=1024, T=512, IN=64, H=128, OUT=2
// Grid: 64 workgroups x 256 threads (8 waves). Each WG owns 16 batch rows and
// runs the full T=512 recurrence for both layers with weights held in VGPRs
// as WMMA B-fragments. h-state is exchanged via DOUBLE-BUFFERED LDS (bf16) so
// each of the two layer phases needs only one workgroup barrier (2/step
// instead of 4 -- the recurrence is latency-bound, barriers are serial cost).
// Gate nonlinearities use native gfx1250 transcendentals (v_exp_f32/v_rcp_f32
// /v_tanh_f32) to keep inner-loop register pressure below the spill threshold.
// ---------------------------------------------------------------------------

typedef __attribute__((ext_vector_type(16))) __bf16 v16bf;
typedef __attribute__((ext_vector_type(8)))  float  v8f;

#define BB   1024
#define TT   512
#define INF  64
#define HH   128

union BFrag {
  v16bf v;
  unsigned short s[16];
  uint4 q[2];
};

__device__ __forceinline__ unsigned short f2bf(float f) {
  union { float f; unsigned u; } x; x.f = f;
  unsigned u = x.u + 0x7FFFu + ((x.u >> 16) & 1u);  // round-to-nearest-even
  return (unsigned short)(u >> 16);
}

#define LOG2E  1.4426950408889634f

__device__ __forceinline__ float fast_sigmoid(float x) {
#if __has_builtin(__builtin_amdgcn_exp2f) && __has_builtin(__builtin_amdgcn_rcpf)
  const float e = __builtin_amdgcn_exp2f(-LOG2E * x);   // v_exp_f32
  return __builtin_amdgcn_rcpf(1.0f + e);               // v_rcp_f32
#else
  return 1.0f / (1.0f + __expf(-x));
#endif
}

__device__ __forceinline__ float fast_tanh(float x) {
#if __has_builtin(__builtin_amdgcn_tanhf)
  return __builtin_amdgcn_tanhf(x);                     // v_tanh_f32 (gfx1250)
#elif __has_builtin(__builtin_amdgcn_exp2f) && __has_builtin(__builtin_amdgcn_rcpf)
  // tanh(x) = 1 - 2/(e^{2x}+1)
  const float e = __builtin_amdgcn_exp2f((2.0f * LOG2E) * x);
  return 1.0f - 2.0f * __builtin_amdgcn_rcpf(e + 1.0f);
#else
  return tanhf(x);
#endif
}

// B-fragment (32x16, K x N) of w^T where w is row-major [3H, Kdim] (so
// B[k][n] = w[n][k], k contiguous in memory -> two 8-float runs per lane, 16 apart).
// Layout: col = lane&15 ; lane<16 holds K=kt*32+{0..7,16..23}, lane>=16 holds K=+{8..15,24..31}.
__device__ __forceinline__ v16bf load_w_frag(const float* __restrict__ w, int Kdim,
                                             int ntile, int kt, int lane) {
  const int col   = ntile * 16 + (lane & 15);
  const int kbase = kt * 32 + ((lane & 16) ? 8 : 0);
  const float* p = w + col * Kdim + kbase;
  BFrag f;
#pragma unroll
  for (int i = 0; i < 16; ++i) f.s[i] = f2bf(p[i < 8 ? i : i + 8]);
  return f.v;
}

// A-fragment (16x32 bf16) from an LDS row-major bf16 buffer.
// lane<16: row=lane, K chunks [0..7] and [16..23]; lane>=16: row=lane-16,
// K chunks [8..15] and [24..31] (per ISA 7.12.2 16-bit A layout).
__device__ __forceinline__ v16bf load_a_frag(const unsigned short* buf, int rowstride,
                                             int kt, int lane) {
  const int row   = lane & 15;
  const int kbase = kt * 32 + ((lane & 16) ? 8 : 0);
  const unsigned short* p = buf + row * rowstride + kbase;
  BFrag f;
  f.q[0] = *(const uint4*)(p);
  f.q[1] = *(const uint4*)(p + 16);
  return f.v;
}

__device__ __forceinline__ v8f splat8(float x) {
  v8f v;
#pragma unroll
  for (int i = 0; i < 8; ++i) v[i] = x;
  return v;
}

#define WMMA_BF16(A, Bf, C) \
  __builtin_amdgcn_wmma_f32_16x16x32_bf16(false, (A), false, (Bf), (short)0, (C), false, false)

__global__ __launch_bounds__(256)
void gru_fused_kernel(const float* __restrict__ x,
                      const float* __restrict__ w_ih0, const float* __restrict__ w_hh0,
                      const float* __restrict__ b_ih0, const float* __restrict__ b_hh0,
                      const float* __restrict__ w_ih1, const float* __restrict__ w_hh1,
                      const float* __restrict__ b_ih1, const float* __restrict__ b_hh1,
                      const float* __restrict__ fc1_w, const float* __restrict__ fc1_b,
                      const float* __restrict__ fc2_w, const float* __restrict__ fc2_b,
                      const float* __restrict__ bn_gamma, const float* __restrict__ bn_beta,
                      const float* __restrict__ bn_mean, const float* __restrict__ bn_var,
                      float* __restrict__ out)
{
  // Double-buffered state (ping-pong so writers never race readers).
  __shared__ __align__(16) unsigned short xb [2][16 * INF];  // x_t tile, bf16
  __shared__ __align__(16) unsigned short h1b[2][16 * HH];   // h1 state, bf16
  __shared__ __align__(16) unsigned short h2b[2][16 * HH];   // h2 state, bf16
  __shared__ __align__(16) float h2f[16 * HH];               // final h2, f32
  __shared__ __align__(16) float o1 [16 * 64];               // fc1 activations

  const int tid    = threadIdx.x;
  const int wv     = tid >> 5;      // wave id 0..7 (= N-tile column group)
  const int lane   = tid & 31;
  const int wgBase = blockIdx.x * 16;

  // ---- Load all weight B-fragments into registers (one-time) ----
  // Wave wv owns N-tiles {wv (r), 8+wv (z), 16+wv (n)} of the 384-wide gates.
  v16bf wih0[3][2], whh0[3][4], wih1[3][4], whh1[3][4];
#pragma unroll
  for (int g = 0; g < 3; ++g) {
    const int nt = g * 8 + wv;
#pragma unroll
    for (int kt = 0; kt < 2; ++kt) wih0[g][kt] = load_w_frag(w_ih0, INF, nt, kt, lane);
#pragma unroll
    for (int kt = 0; kt < 4; ++kt) {
      whh0[g][kt] = load_w_frag(w_hh0, HH, nt, kt, lane);
      wih1[g][kt] = load_w_frag(w_ih1, HH, nt, kt, lane);
      whh1[g][kt] = load_w_frag(w_hh1, HH, nt, kt, lane);
    }
  }

  // This wave's h-column for this lane (matches C-fragment N = lane&15).
  const int c   = lane & 15;
  const int col = wv * 16 + c;

  // Per-lane gate biases (constant over time).
  const float br0  = b_ih0[col]          + b_hh0[col];
  const float bz0  = b_ih0[HH + col]     + b_hh0[HH + col];
  const float bxn0 = b_ih0[2 * HH + col];
  const float bhn0 = b_hh0[2 * HH + col];
  const float br1  = b_ih1[col]          + b_hh1[col];
  const float bz1  = b_ih1[HH + col]     + b_hh1[HH + col];
  const float bxn1 = b_ih1[2 * HH + col];
  const float bhn1 = b_hh1[2 * HH + col];

  // h states: f32 copies live in registers (C-fragment layout), bf16 in LDS.
  v8f h1o = splat8(0.0f), h2o = splat8(0.0f);
  for (int i = tid; i < 16 * HH; i += 256) { h1b[0][i] = 0; h2b[0][i] = 0; }

  // Stage x_0 (16 rows x 64 cols, f32 -> bf16), 4 elems/thread, coalesced rows.
  const int sm = tid >> 4, skb = (tid & 15) * 4;   // staging coords
  const float* xrow = x + (size_t)(wgBase + sm) * TT * INF + skb;
  {
    const float4 v = *(const float4*)(xrow);
    xb[0][sm * INF + skb + 0] = f2bf(v.x);
    xb[0][sm * INF + skb + 1] = f2bf(v.y);
    xb[0][sm * INF + skb + 2] = f2bf(v.z);
    xb[0][sm * INF + skb + 3] = f2bf(v.w);
  }
  __syncthreads();

  const int mh = (lane & 16) ? 8 : 0;  // C-fragment row base

#pragma unroll 1
  for (int t = 0; t < TT; ++t) {
    const int cur = t & 1;
    const int nxt = cur ^ 1;

    // ---------------- layer 0 : reads xb[cur], h1b[cur] ----------------
    v8f ar = splat8(br0), az = splat8(bz0), axn = splat8(bxn0), ahn = splat8(bhn0);
#pragma unroll
    for (int kt = 0; kt < 2; ++kt) {               // x_t @ w_ih0^T
      v16bf a = load_a_frag(xb[cur], INF, kt, lane);
      ar  = WMMA_BF16(a, wih0[0][kt], ar);
      az  = WMMA_BF16(a, wih0[1][kt], az);
      axn = WMMA_BF16(a, wih0[2][kt], axn);
    }
#pragma unroll
    for (int kt = 0; kt < 4; ++kt) {               // h1_{t-1} @ w_hh0^T
      v16bf a = load_a_frag(h1b[cur], HH, kt, lane);
      ar  = WMMA_BF16(a, whh0[0][kt], ar);
      az  = WMMA_BF16(a, whh0[1][kt], az);
      ahn = WMMA_BF16(a, whh0[2][kt], ahn);
    }
    // GRU elementwise; publish new h1 into the OTHER buffer (no race).
#pragma unroll
    for (int d = 0; d < 8; ++d) {
      const float r = fast_sigmoid(ar[d]);
      const float z = fast_sigmoid(az[d]);
      const float n = fast_tanh(axn[d] + r * ahn[d]);
      const float hnew = (1.0f - z) * n + z * h1o[d];
      h1o[d] = hnew;
      h1b[nxt][(mh + d) * HH + col] = f2bf(hnew);
    }
    // Stage x_{t+1} into xb[nxt]; prefetch x_{t+2} into cache.
    if (t + 1 < TT) {
      const float4 v = *(const float4*)(xrow + (size_t)(t + 1) * INF);
      xb[nxt][sm * INF + skb + 0] = f2bf(v.x);
      xb[nxt][sm * INF + skb + 1] = f2bf(v.y);
      xb[nxt][sm * INF + skb + 2] = f2bf(v.z);
      xb[nxt][sm * INF + skb + 3] = f2bf(v.w);
    }
    if (t + 2 < TT) {
      __builtin_prefetch(xrow + (size_t)(t + 2) * INF, 0, 3);  // global_prefetch_b8
    }
    __syncthreads();   // barrier 1: h1_t + x_{t+1} visible

    // ---------------- layer 1 : reads h1b[nxt] (=h1_t), h2b[cur] ----------------
    ar = splat8(br1); az = splat8(bz1); axn = splat8(bxn1); ahn = splat8(bhn1);
#pragma unroll
    for (int kt = 0; kt < 4; ++kt) {               // h1_t @ w_ih1^T
      v16bf a = load_a_frag(h1b[nxt], HH, kt, lane);
      ar  = WMMA_BF16(a, wih1[0][kt], ar);
      az  = WMMA_BF16(a, wih1[1][kt], az);
      axn = WMMA_BF16(a, wih1[2][kt], axn);
    }
#pragma unroll
    for (int kt = 0; kt < 4; ++kt) {               // h2_{t-1} @ w_hh1^T
      v16bf a = load_a_frag(h2b[cur], HH, kt, lane);
      ar  = WMMA_BF16(a, whh1[0][kt], ar);
      az  = WMMA_BF16(a, whh1[1][kt], az);
      ahn = WMMA_BF16(a, whh1[2][kt], ahn);
    }
#pragma unroll
    for (int d = 0; d < 8; ++d) {
      const float r = fast_sigmoid(ar[d]);
      const float z = fast_sigmoid(az[d]);
      const float n = fast_tanh(axn[d] + r * ahn[d]);
      const float hnew = (1.0f - z) * n + z * h2o[d];
      h2o[d] = hnew;
      h2b[nxt][(mh + d) * HH + col] = f2bf(hnew);
    }
    __syncthreads();   // barrier 2: h2_t visible for next step
  }

  // ---- Epilogue: fc1 -> relu -> bn -> fc2 on the final h2 (f32 precision) ----
#pragma unroll
  for (int d = 0; d < 8; ++d) h2f[(mh + d) * HH + col] = h2o[d];
  __syncthreads();

#pragma unroll 1
  for (int e = 0; e < 4; ++e) {                    // 16x64 fc1 outputs
    const int o = tid * 4 + e;
    const int m = o >> 6, j = o & 63;
    float s = fc1_b[j];
    for (int k = 0; k < HH; ++k) s += fc1_w[j * HH + k] * h2f[m * HH + k];
    s = fmaxf(s, 0.0f);
    s = (s - bn_mean[j]) * rsqrtf(bn_var[j] + 1e-5f) * bn_gamma[j] + bn_beta[j];
    o1[m * 64 + j] = s;
  }
  __syncthreads();

  if (tid < 32) {                                  // 16x2 fc2 outputs
    const int m = tid >> 1, o = tid & 1;
    float s = fc2_b[o];
    for (int k = 0; k < 64; ++k) s += fc2_w[o * 64 + k] * o1[m * 64 + k];
    out[(size_t)(wgBase + m) * 2 + o] = s;
  }
}

extern "C" void kernel_launch(void* const* d_in, const int* in_sizes, int n_in,
                              void* d_out, int out_size, void* d_ws, size_t ws_size,
                              hipStream_t stream) {
  (void)in_sizes; (void)n_in; (void)out_size; (void)d_ws; (void)ws_size;
  const float* x        = (const float*)d_in[0];
  const float* w_ih0    = (const float*)d_in[1];
  const float* w_hh0    = (const float*)d_in[2];
  const float* b_ih0    = (const float*)d_in[3];
  const float* b_hh0    = (const float*)d_in[4];
  const float* w_ih1    = (const float*)d_in[5];
  const float* w_hh1    = (const float*)d_in[6];
  const float* b_ih1    = (const float*)d_in[7];
  const float* b_hh1    = (const float*)d_in[8];
  const float* fc1_w    = (const float*)d_in[9];
  const float* fc1_b    = (const float*)d_in[10];
  const float* fc2_w    = (const float*)d_in[11];
  const float* fc2_b    = (const float*)d_in[12];
  const float* bn_gamma = (const float*)d_in[13];
  const float* bn_beta  = (const float*)d_in[14];
  const float* bn_mean  = (const float*)d_in[15];
  const float* bn_var   = (const float*)d_in[16];

  dim3 grid(BB / 16), block(256);
  gru_fused_kernel<<<grid, block, 0, stream>>>(
      x, w_ih0, w_hh0, b_ih0, b_hh0, w_ih1, w_hh1, b_ih1, b_hh1,
      fc1_w, fc1_b, fc2_w, fc2_b, bn_gamma, bn_beta, bn_mean, bn_var,
      (float*)d_out);
}
